// TransformerSetAbstraction_39213051412828
// MI455X (gfx1250) — matrix operand order repaired and checked by reading, not compiled, not run on hardware
//
#include <hip/hip_runtime.h>
#include <hip/hip_bf16.h>
#include <stdint.h>

typedef __bf16 bf16;
typedef __attribute__((ext_vector_type(16))) __bf16 v16bf;
typedef __attribute__((ext_vector_type(8)))  float  v8f;

#define BB   8
#define NPTS 8192
#define NP   2048
#define NN   16
#define DD   256
#define BROWS (BB*NP)          // 16384 rows for the global BatchNorms

union FragU { v16bf v; uint4 u[2]; };

// ---------------------------------------------------------------------------
// f32 -> bf16 conversion
// ---------------------------------------------------------------------------
__global__ void cvt_bf16_kernel(const float* __restrict__ in, bf16* __restrict__ out, int n) {
  int i = blockIdx.x * blockDim.x + threadIdx.x;
  if (i < n) out[i] = (bf16)in[i];
}

// ---------------------------------------------------------------------------
// WMMA GEMM:  Y[M,256] = epilogue( X[M,256] @ W[256,256]^T )
// One wave computes a 16x32 output strip (two 16x16 N-tiles, shared A frag).
// A/B fragments use contiguous k: lane half h holds k 16h..16h+15, same in A and B (16-bit A 16x32, B 32x16).
// ---------------------------------------------------------------------------
__global__ void gemm256_kernel(const bf16* __restrict__ A, const bf16* __restrict__ W,
                               const float* __restrict__ bias, const float* __restrict__ resid,
                               bf16* __restrict__ outb, float* __restrict__ outf,
                               int mtiles, int relu)
{
  int wave = threadIdx.x >> 5;
  int lane = threadIdx.x & 31;
  int pair = blockIdx.x * (blockDim.x >> 5) + wave;   // one pair = 16x32 strip
  int npairs = mtiles << 3;                            // 8 N-pairs per M-tile
  if (pair >= npairs) return;
  int mt  = pair >> 3;
  int nt0 = (pair & 7) << 1;
  int half = lane >> 4;      // lane half selects K sub-range
  int l15  = lane & 15;

  const bf16* arow  = A + (size_t)(mt * 16 + l15) * DD + (half ? 16 : 0);
  const bf16* brow0 = W + (size_t)(nt0 * 16 + l15) * DD + (half ? 16 : 0);
  const bf16* brow1 = brow0 + 16 * DD;

  v8f acc0 = {}; v8f acc1 = {};
  #pragma unroll
  for (int k0 = 0; k0 < DD; k0 += 32) {
    FragU a, b0, b1;
    // A 16x32 bf16: lanes0-15 rows M, K {k0..k0+15}; lanes16-31 K {k0+16..k0+31}
    a.u[0]  = *(const uint4*)(arow + k0);
    a.u[1]  = *(const uint4*)(arow + k0 + 8);
    // B 32x16 bf16: lanes0-15 col N, K {k0..k0+15}; lanes16-31 K {k0+16..k0+31}
    b0.u[0] = *(const uint4*)(brow0 + k0);
    b0.u[1] = *(const uint4*)(brow0 + k0 + 8);
    b1.u[0] = *(const uint4*)(brow1 + k0);
    b1.u[1] = *(const uint4*)(brow1 + k0 + 8);
    acc0 = __builtin_amdgcn_wmma_f32_16x16x32_bf16(false, a.v, false, b0.v, (short)0, acc0, false, false);
    acc1 = __builtin_amdgcn_wmma_f32_16x16x32_bf16(false, a.v, false, b1.v, (short)0, acc1, false, false);
  }

  #pragma unroll
  for (int t = 0; t < 2; ++t) {
    int col = (nt0 + t) * 16 + l15;
    float bb = bias ? bias[col] : 0.0f;
    #pragma unroll
    for (int r = 0; r < 8; ++r) {
      int row = mt * 16 + r + (half ? 8 : 0);
      float x = (t ? acc1[r] : acc0[r]) + bb;
      size_t o = (size_t)row * DD + col;
      if (resid) x += resid[o];
      if (relu)  x = x > 0.f ? x : 0.f;
      if (outf) outf[o] = x;
      if (outb) outb[o] = (bf16)x;
    }
  }
}

// ---------------------------------------------------------------------------
// gather rows of a bf16 matrix:  dst[m,:] = src[idx[m],:]
// ---------------------------------------------------------------------------
__global__ void gather_rows_kernel(const bf16* __restrict__ src, const int* __restrict__ idx,
                                   bf16* __restrict__ dst, int rows)
{
  int i = blockIdx.x * blockDim.x + threadIdx.x;
  if (i >= rows * DD) return;
  int m = i >> 8, f = i & 255;
  dst[i] = src[(size_t)idx[m] * DD + f];
}

// relpos[m*NN+n][3] = xyz[knn[m,n]] - xyz[fps[m]]
__global__ void relpos_kernel(const float* __restrict__ xyz, const int* __restrict__ fps,
                              const int* __restrict__ knn, float* __restrict__ relpos)
{
  int r = blockIdx.x * blockDim.x + threadIdx.x;
  if (r >= NP * NN) return;
  int m = r >> 4;
  int c = fps[m];
  int j = knn[r];
  #pragma unroll
  for (int d = 0; d < 3; ++d)
    relpos[(size_t)r * 3 + d] = xyz[(size_t)j * 3 + d] - xyz[(size_t)c * 3 + d];
}

// pos hidden: relu(relpos @ dw1^T + db1)  (K=3, scalar FMA)
__global__ void pos_hidden_kernel(const float* __restrict__ relpos, const float* __restrict__ dw1,
                                  const float* __restrict__ db1, bf16* __restrict__ out)
{
  int i = blockIdx.x * blockDim.x + threadIdx.x;
  if (i >= NP * NN * DD) return;
  int r = i >> 8, f = i & 255;
  float h = db1[f] + relpos[(size_t)r*3]   * dw1[(size_t)f*3]
                   + relpos[(size_t)r*3+1] * dw1[(size_t)f*3+1]
                   + relpos[(size_t)r*3+2] * dw1[(size_t)f*3+2];
  out[i] = (bf16)(h > 0.f ? h : 0.f);
}

// t[m,n,f] = q[m,f] - Kp[knn[m,n],f] + pos[m,n,f]
__global__ void form_t_kernel(const bf16* __restrict__ Q, const bf16* __restrict__ Kp,
                              const bf16* __restrict__ Pos, const int* __restrict__ knn,
                              bf16* __restrict__ out)
{
  int i = blockIdx.x * blockDim.x + threadIdx.x;
  if (i >= NP * NN * DD) return;
  int r = i >> 8, f = i & 255;
  int m = r >> 4;
  int j = knn[r];
  float t = (float)Q[(size_t)m * DD + f] - (float)Kp[(size_t)j * DD + f] + (float)Pos[i];
  out[i] = (bf16)t;
}

// per (m,f): softmax over n of S[m,n,f]; res = sum_n a*(Vp[knn]+pos)
__global__ void attn_reduce_kernel(const float* __restrict__ S, const bf16* __restrict__ Pos,
                                   const bf16* __restrict__ Vp, const int* __restrict__ knn,
                                   float* __restrict__ resf, bf16* __restrict__ resb)
{
  int i = blockIdx.x * blockDim.x + threadIdx.x;
  if (i >= NP * DD) return;
  int m = i >> 8, f = i & 255;
  float s[NN];
  float mx = -3.0e38f;
  #pragma unroll
  for (int n = 0; n < NN; ++n) {
    s[n] = S[((size_t)(m * NN + n)) * DD + f];
    mx = fmaxf(mx, s[n]);
  }
  float den = 0.f;
  #pragma unroll
  for (int n = 0; n < NN; ++n) { s[n] = __expf(s[n] - mx); den += s[n]; }
  float inv = 1.f / den, acc = 0.f;
  #pragma unroll
  for (int n = 0; n < NN; ++n) {
    int r = m * NN + n;
    int j = knn[r];
    float v = (float)Vp[(size_t)j * DD + f] + (float)Pos[(size_t)r * DD + f];
    acc += s[n] * inv * v;
  }
  resf[i] = acc;
  if (resb) resb[i] = (bf16)acc;
}

// ---------------------------------------------------------------------------
// BatchNorm over BROWS rows, 256 channels: deterministic two-stage reduction
// ---------------------------------------------------------------------------
__global__ void bn_part_kernel(const float* __restrict__ X, float* __restrict__ part)
{
  int g = blockIdx.x;     // 64 groups of 256 rows
  int f = threadIdx.x;    // channel
  float s = 0.f, ss = 0.f;
  int r0 = g * (BROWS / 64);
  for (int r = 0; r < BROWS / 64; ++r) {
    float x = X[(size_t)(r0 + r) * DD + f];
    s += x; ss += x * x;
  }
  part[(size_t)g * 2 * DD + f]      = s;
  part[(size_t)g * 2 * DD + DD + f] = ss;
}

__global__ void bn_stats_kernel(const float* __restrict__ part, float* __restrict__ stats)
{
  int f = threadIdx.x;
  float s = 0.f, ss = 0.f;
  for (int g = 0; g < 64; ++g) {
    s  += part[(size_t)g * 2 * DD + f];
    ss += part[(size_t)g * 2 * DD + DD + f];
  }
  float mu  = s / (float)BROWS;
  float var = ss / (float)BROWS - mu * mu;
  stats[f]      = mu;
  stats[DD + f] = rsqrtf(var + 1e-5f);
}

__global__ void bn_apply_kernel(const float* __restrict__ X, const float* __restrict__ stats,
                                const float* __restrict__ g, const float* __restrict__ b,
                                float* __restrict__ outf, bf16* __restrict__ outb, int relu)
{
  int i = blockIdx.x * blockDim.x + threadIdx.x;
  if (i >= BROWS * DD) return;
  int f = i & 255;
  float y = (X[i] - stats[f]) * stats[DD + f] * g[f] + b[f];
  if (relu) y = y > 0.f ? y : 0.f;
  if (outf) outf[i] = y;
  if (outb) outb[i] = (bf16)y;
}

// ---------------------------------------------------------------------------
// misc elementwise
// ---------------------------------------------------------------------------
__global__ void add2_kernel(const float* __restrict__ A, float* __restrict__ B_, int n) {
  int i = blockIdx.x * blockDim.x + threadIdx.x;
  if (i < n) B_[i] += A[i];
}

__global__ void add_gather_points_kernel(const float* __restrict__ points,
                                         const int* __restrict__ fps, float* __restrict__ X)
{
  int i = blockIdx.x * blockDim.x + threadIdx.x;
  if (i >= BROWS * DD) return;
  int row = i >> 8, f = i & 255;
  int b = row >> 11, m = row & (NP - 1);
  int j = fps[b * NP + m];
  X[i] += points[((size_t)b * NPTS + j) * DD + f];
}

__global__ void new_xyz_kernel(const float* __restrict__ xyz, const int* __restrict__ fps,
                               float* __restrict__ out)
{
  int i = blockIdx.x * blockDim.x + threadIdx.x;
  if (i >= BB * NP * 3) return;
  int p = i / 3, d = i - p * 3;
  int b = p >> 11, m = p & (NP - 1);
  int j = fps[b * NP + m];
  out[i] = xyz[((size_t)b * NPTS + j) * 3 + d];
}

// ---------------------------------------------------------------------------
static inline int ceildiv(int a, int b) { return (a + b - 1) / b; }

static void launch_gemm(const bf16* A, const bf16* W, const float* bias, const float* resid,
                        bf16* outb, float* outf, int mtiles, int relu, hipStream_t s)
{
  int pairs = mtiles * 8;
  gemm256_kernel<<<ceildiv(pairs, 8), 256, 0, s>>>(A, W, bias, resid, outb, outf, mtiles, relu);
}

extern "C" void kernel_launch(void* const* d_in, const int* in_sizes, int n_in,
                              void* d_out, int out_size, void* d_ws, size_t ws_size,
                              hipStream_t stream)
{
  (void)in_sizes; (void)n_in; (void)out_size;
  const float* xyz    = (const float*)d_in[0];
  const float* points = (const float*)d_in[1];
  const int*   fps    = (const int*)d_in[2];
  const int*   knn    = (const int*)d_in[3];
  const float* wsrc[13] = {
    (const float*)d_in[4],  // wq
    (const float*)d_in[5],  // wk
    (const float*)d_in[6],  // wv
    (const float*)d_in[7],  // wq2
    (const float*)d_in[8],  // wk2
    (const float*)d_in[9],  // wv2
    (const float*)d_in[12], // dw2
    (const float*)d_in[14], // g1w1
    (const float*)d_in[16], // g1w2
    (const float*)d_in[18], // g2w1
    (const float*)d_in[20], // g2w2
    (const float*)d_in[22], // c1w
    (const float*)d_in[24], // c2w
  };
  enum { IWQ, IWK, IWV, IWQ2, IWK2, IWV2, IDW2, IG1W1, IG1W2, IG2W1, IG2W2, IC1W, IC2W };
  const float* dw1  = (const float*)d_in[10];
  const float* db1  = (const float*)d_in[11];
  const float* db2  = (const float*)d_in[13];
  const float* g1b1 = (const float*)d_in[15];
  const float* g1b2 = (const float*)d_in[17];
  const float* g2b1 = (const float*)d_in[19];
  const float* g2b2 = (const float*)d_in[21];
  const float* c1b  = (const float*)d_in[23];
  const float* c2b  = (const float*)d_in[25];
  const float* bn1g = (const float*)d_in[26];
  const float* bn1b = (const float*)d_in[27];
  const float* bn0g = (const float*)d_in[28];
  const float* bn0b = (const float*)d_in[29];
  const float* bnAg = (const float*)d_in[30];
  const float* bnAb = (const float*)d_in[31];
  const float* bnBg = (const float*)d_in[32];
  const float* bnBb = (const float*)d_in[33];

  // ---- carve workspace -----------------------------------------------------
  uint8_t* base = (uint8_t*)d_ws;
  size_t off = 0;
  auto carve = [&](size_t bytes) -> void* {
    void* p = base + off;
    off += (bytes + 255) & ~(size_t)255;
    return p;
  };
  bf16* wbf[13];
  for (int i = 0; i < 13; ++i) wbf[i] = (bf16*)carve((size_t)DD * DD * sizeof(bf16));
  bf16*  Pbf      = (bf16*) carve((size_t)NPTS * DD * 2);
  bf16*  Kp       = (bf16*) carve((size_t)NPTS * DD * 2);
  bf16*  Vp       = (bf16*) carve((size_t)NPTS * DD * 2);
  bf16*  Qg       = (bf16*) carve((size_t)NP * DD * 2);
  bf16*  Qp       = (bf16*) carve((size_t)NP * DD * 2);
  float* relpos   = (float*)carve((size_t)NP * NN * 3 * 4);
  bf16*  PosB     = (bf16*) carve((size_t)NP * NN * DD * 2);
  bf16*  G1       = (bf16*) carve((size_t)NP * NN * DD * 2);
  bf16*  G2       = (bf16*) carve((size_t)NP * NN * DD * 2);
  float* Sf32     = (float*)carve((size_t)NP * NN * DD * 4);
  float* res1_pre = (float*)carve((size_t)BROWS * DD * 4);
  bf16*  res1_bf  = (bf16*) carve((size_t)BROWS * DD * 2);
  float* Hc       = (float*)carve((size_t)BROWS * DD * 4);
  bf16*  Hc_bf    = (bf16*) carve((size_t)BROWS * DD * 2);
  float* res1n    = (float*)carve((size_t)BROWS * DD * 4);
  bf16*  res1n_bf = (bf16*) carve((size_t)BROWS * DD * 2);
  bf16*  q2bf     = (bf16*) carve((size_t)BROWS * DD * 2);
  float* res2     = (float*)carve((size_t)BROWS * DD * 4);
  float* part     = (float*)carve((size_t)64 * 2 * DD * 4);
  float* stats    = (float*)carve((size_t)2 * DD * 4);
  if (off > ws_size) return;  // workspace too small: do nothing (deterministic)

  const int ELT = 256;
  const int nPNN = NP * NN * DD;   // 8.4M
  const int nPD  = NP * DD;        // 0.52M
  const int nBR  = BROWS * DD;     // 4.2M

  // ---- convert weights to bf16 --------------------------------------------
  for (int i = 0; i < 13; ++i)
    cvt_bf16_kernel<<<ceildiv(DD * DD, ELT), ELT, 0, stream>>>(wsrc[i], wbf[i], DD * DD);

  // ---- phase 1: attention branch 1, per batch -----------------------------
  for (int b = 0; b < BB; ++b) {
    const int* fpsb = fps + b * NP;
    const int* knnb = knn + (size_t)b * NP * NN;
    cvt_bf16_kernel<<<ceildiv(NPTS * DD, ELT), ELT, 0, stream>>>(
        points + (size_t)b * NPTS * DD, Pbf, NPTS * DD);
    launch_gemm(Pbf, wbf[IWK], nullptr, nullptr, Kp, nullptr, NPTS / 16, 0, stream);
    launch_gemm(Pbf, wbf[IWV], nullptr, nullptr, Vp, nullptr, NPTS / 16, 0, stream);
    gather_rows_kernel<<<ceildiv(nPD, ELT), ELT, 0, stream>>>(Pbf, fpsb, Qg, NP);
    launch_gemm(Qg, wbf[IWQ], nullptr, nullptr, Qp, nullptr, NP / 16, 0, stream);
    relpos_kernel<<<ceildiv(NP * NN, ELT), ELT, 0, stream>>>(
        xyz + (size_t)b * NPTS * 3, fpsb, knnb, relpos);
    pos_hidden_kernel<<<ceildiv(nPNN, ELT), ELT, 0, stream>>>(relpos, dw1, db1, G1);
    launch_gemm(G1, wbf[IDW2], db2, nullptr, PosB, nullptr, NP * NN / 16, 0, stream);
    form_t_kernel<<<ceildiv(nPNN, ELT), ELT, 0, stream>>>(Qp, Kp, PosB, knnb, G1);
    launch_gemm(G1, wbf[IG1W1], g1b1, nullptr, G2, nullptr, NP * NN / 16, 1, stream);
    launch_gemm(G2, wbf[IG1W2], g1b2, nullptr, nullptr, Sf32, NP * NN / 16, 0, stream);
    attn_reduce_kernel<<<ceildiv(nPD, ELT), ELT, 0, stream>>>(
        Sf32, PosB, Vp, knnb, res1_pre + (size_t)b * NP * DD, res1_bf + (size_t)b * NP * DD);
  }

  // ---- global: conv block + BN1 + BN0 + q2 projection ---------------------
  launch_gemm(res1_bf, wbf[IC1W], c1b, nullptr, nullptr, Hc, BROWS / 16, 0, stream);
  bn_part_kernel<<<64, 256, 0, stream>>>(Hc, part);
  bn_stats_kernel<<<1, 256, 0, stream>>>(part, stats);
  bn_apply_kernel<<<ceildiv(nBR, ELT), ELT, 0, stream>>>(Hc, stats, bn1g, bn1b, nullptr, Hc_bf, 1);
  launch_gemm(Hc_bf, wbf[IC2W], c2b, res1_pre, nullptr, res1_pre, BROWS / 16, 0, stream);
  bn_part_kernel<<<64, 256, 0, stream>>>(res1_pre, part);
  bn_stats_kernel<<<1, 256, 0, stream>>>(part, stats);
  bn_apply_kernel<<<ceildiv(nBR, ELT), ELT, 0, stream>>>(res1_pre, stats, bn0g, bn0b, res1n, res1n_bf, 0);
  launch_gemm(res1n_bf, wbf[IWQ2], nullptr, nullptr, q2bf, nullptr, BROWS / 16, 0, stream);

  // ---- phase 2: attention branch 2, per batch -----------------------------
  for (int b = 0; b < BB; ++b) {
    const int* fpsb = fps + b * NP;
    const int* knnb = knn + (size_t)b * NP * NN;
    cvt_bf16_kernel<<<ceildiv(NPTS * DD, ELT), ELT, 0, stream>>>(
        points + (size_t)b * NPTS * DD, Pbf, NPTS * DD);
    launch_gemm(Pbf, wbf[IWK2], nullptr, nullptr, Kp, nullptr, NPTS / 16, 0, stream);
    launch_gemm(Pbf, wbf[IWV2], nullptr, nullptr, Vp, nullptr, NPTS / 16, 0, stream);
    relpos_kernel<<<ceildiv(NP * NN, ELT), ELT, 0, stream>>>(
        xyz + (size_t)b * NPTS * 3, fpsb, knnb, relpos);
    pos_hidden_kernel<<<ceildiv(nPNN, ELT), ELT, 0, stream>>>(relpos, dw1, db1, G1);
    launch_gemm(G1, wbf[IDW2], db2, nullptr, PosB, nullptr, NP * NN / 16, 0, stream);
    form_t_kernel<<<ceildiv(nPNN, ELT), ELT, 0, stream>>>(
        q2bf + (size_t)b * NP * DD, Kp, PosB, knnb, G1);
    launch_gemm(G1, wbf[IG2W1], g2b1, nullptr, G2, nullptr, NP * NN / 16, 1, stream);
    launch_gemm(G2, wbf[IG2W2], g2b2, nullptr, nullptr, Sf32, NP * NN / 16, 0, stream);
    attn_reduce_kernel<<<ceildiv(nPD, ELT), ELT, 0, stream>>>(
        Sf32, PosB, Vp, knnb, res2 + (size_t)b * NP * DD, nullptr);
  }

  // ---- final: BNA, + gathered points, BNB, outputs ------------------------
  add2_kernel<<<ceildiv(nBR, ELT), ELT, 0, stream>>>(res1n, res2, nBR);
  bn_part_kernel<<<64, 256, 0, stream>>>(res2, part);
  bn_stats_kernel<<<1, 256, 0, stream>>>(part, stats);
  bn_apply_kernel<<<ceildiv(nBR, ELT), ELT, 0, stream>>>(res2, stats, bnAg, bnAb, res2, nullptr, 0);
  add_gather_points_kernel<<<ceildiv(nBR, ELT), ELT, 0, stream>>>(points, fps, res2);
  bn_part_kernel<<<64, 256, 0, stream>>>(res2, part);
  bn_stats_kernel<<<1, 256, 0, stream>>>(part, stats);
  float* out_np = (float*)d_out + (size_t)BB * NP * 3;
  bn_apply_kernel<<<ceildiv(nBR, ELT), ELT, 0, stream>>>(res2, stats, bnBg, bnBb, out_np, nullptr, 0);
  new_xyz_kernel<<<ceildiv(BB * NP * 3, ELT), ELT, 0, stream>>>(xyz, fps, (float*)d_out);
}
